// PFNetTransformer_75900662054931
// MI455X (gfx1250) — compile-verified, matrix operand order checked by reading, not run on hardware
//
#include <hip/hip_runtime.h>

// ---------------------------------------------------------------------------
// PFNet transformer forward for MI455X (gfx1250, wave32, WMMA).
// All heavy contractions run through v_wmma_f32_16x16x32_bf16 with f32 accum.
// ---------------------------------------------------------------------------

#define MTOK 16384      // B*N tokens
#define NN   4096       // tokens per batch
#define DD   128        // model dim
#define HH   8          // heads
#define HK   1024       // H*K

typedef __bf16 bf16;
typedef __attribute__((ext_vector_type(8)))  bf16  v8bf;
typedef __attribute__((ext_vector_type(16))) bf16  v16bf;
typedef __attribute__((ext_vector_type(8)))  float v8f;

enum { EPI_LIN = 0, EPI_ELU = 1, EPI_QP = 2, EPI_KP = 3, EPI_SCAT = 4, EPI_MASK = 5 };

__device__ __forceinline__ float eluf(float x) { return x > 0.f ? x : (expf(x) - 1.f); }

#define TM 128
#define TN 64
#define TKS 32

__device__ __forceinline__ v8bf cvt8(const float* f) {
  v8bf v;
  #pragma unroll
  for (int i = 0; i < 8; ++i) v[i] = (bf16)f[i];
  return v;
}

// Fragment per ISA 7.12.2 16-bit layout: lane L -> row = L%16,
// K = (e<8?0:16) + (L/16)*8 + e%8  => two contiguous 16B LDS loads.
__device__ __forceinline__ v16bf ldfrag(const bf16* row, int hl) {
  v8bf lo = *(const v8bf*)(row + hl * 8);
  v8bf hi = *(const v8bf*)(row + 16 + hl * 8);
  return __builtin_shufflevector(lo, hi, 0,1,2,3,4,5,6,7,8,9,10,11,12,13,14,15);
}

// ----------------------------- WMMA GEMM -----------------------------------
// C(M,N) = act(A(M,K) * B(K,N) + bias).  TRANSA: A addr = k*lda + m.
// Requirements: M % TM == 0 at every call site (staging is unguarded in M).
// Batched via blockIdx.z with element strides. 128x64 tile, 8 waves, each
// wave computes a 32x32 block = 4 WMMAs per 32-deep K step. Register
// double-buffering overlaps next-slab global loads with current WMMAs.
template<int TRANSA, int EPI>
__global__ __launch_bounds__(256) void wmma_gemm(
    const float* __restrict__ A, const float* __restrict__ Bw,
    const float* __restrict__ bias, float* __restrict__ Cp,
    const float* __restrict__ mask,
    int Mtot, int Ntot, int Ktrue, int lda,
    long sA, long sB, long sC)
{
  A  += (long)blockIdx.z * sA;
  Bw += (long)blockIdx.z * sB;
  Cp += (long)blockIdx.z * sC;

  __shared__ __attribute__((aligned(16))) bf16 As[TM][TKS];   // 8 KB
  __shared__ __attribute__((aligned(16))) bf16 BsT[TN][TKS];  // 4 KB (B transposed: [n][k])

  const int tid  = threadIdx.x;
  const int wave = tid >> 5;
  const int lane = tid & 31;
  const int lm   = lane & 15;
  const int hl   = lane >> 4;
  const int wm   = (wave >> 1) * 32;   // wave M block (4 groups of 32)
  const int wn   = (wave & 1) * 32;    // wave N block (2 groups of 32)
  const int tileM0 = blockIdx.y * TM;
  const int tileN0 = blockIdx.x * TN;

  // staging thread mappings
  const int a_r  = tid >> 1;           // TRANSA=0: row, 16-wide K group
  const int a_c0 = (tid & 1) * 16;
  const int a_kk = tid >> 3;           // TRANSA=1: k row, 16-wide M group
  const int a_m0 = (tid & 7) * 16;
  const int b_kk = tid >> 3;           // B: k row, 8-wide N group
  const int b_n0 = (tid & 7) * 8;

  v8f zero8 = {0.f,0.f,0.f,0.f,0.f,0.f,0.f,0.f};
  v8f acc[2][2];
  acc[0][0] = zero8; acc[0][1] = zero8; acc[1][0] = zero8; acc[1][1] = zero8;

  float ra[16], rb[8];

  auto loadA = [&](int k0) {
    if (TRANSA == 0) {
      const int gm = tileM0 + a_r, gk0 = k0 + a_c0;
      if (gk0 + 16 <= Ktrue) {
        const float4* p = (const float4*)(A + (long)gm * lda + gk0);
        float4 x0 = p[0], x1 = p[1], x2 = p[2], x3 = p[3];
        ra[0]=x0.x; ra[1]=x0.y; ra[2]=x0.z;  ra[3]=x0.w;
        ra[4]=x1.x; ra[5]=x1.y; ra[6]=x1.z;  ra[7]=x1.w;
        ra[8]=x2.x; ra[9]=x2.y; ra[10]=x2.z; ra[11]=x2.w;
        ra[12]=x3.x; ra[13]=x3.y; ra[14]=x3.z; ra[15]=x3.w;
      } else {
        #pragma unroll
        for (int i = 0; i < 16; ++i) {
          int gk = gk0 + i;
          ra[i] = (gk < Ktrue) ? A[(long)gm * lda + gk] : 0.f;
        }
      }
    } else {
      const int gk = k0 + a_kk, gm0 = tileM0 + a_m0;
      if (gk < Ktrue) {
        const float4* p = (const float4*)(A + (long)gk * lda + gm0);
        float4 x0 = p[0], x1 = p[1], x2 = p[2], x3 = p[3];
        ra[0]=x0.x; ra[1]=x0.y; ra[2]=x0.z;  ra[3]=x0.w;
        ra[4]=x1.x; ra[5]=x1.y; ra[6]=x1.z;  ra[7]=x1.w;
        ra[8]=x2.x; ra[9]=x2.y; ra[10]=x2.z; ra[11]=x2.w;
        ra[12]=x3.x; ra[13]=x3.y; ra[14]=x3.z; ra[15]=x3.w;
      } else {
        #pragma unroll
        for (int i = 0; i < 16; ++i) ra[i] = 0.f;
      }
    }
  };

  auto loadB = [&](int k0) {
    const int gk = k0 + b_kk, gn0 = tileN0 + b_n0;
    if (gk < Ktrue && gn0 + 8 <= Ntot) {
      const float4* p = (const float4*)(Bw + (long)gk * Ntot + gn0);
      float4 x0 = p[0], x1 = p[1];
      rb[0]=x0.x; rb[1]=x0.y; rb[2]=x0.z; rb[3]=x0.w;
      rb[4]=x1.x; rb[5]=x1.y; rb[6]=x1.z; rb[7]=x1.w;
    } else {
      #pragma unroll
      for (int i = 0; i < 8; ++i) {
        int gn = gn0 + i;
        rb[i] = (gk < Ktrue && gn < Ntot) ? Bw[(long)gk * Ntot + gn] : 0.f;
      }
    }
  };

  auto storeTiles = [&]() {
    if (TRANSA == 0) {
      *(v8bf*)&As[a_r][a_c0]     = cvt8(ra);
      *(v8bf*)&As[a_r][a_c0 + 8] = cvt8(ra + 8);
    } else {
      #pragma unroll
      for (int i = 0; i < 16; ++i) As[a_m0 + i][a_kk] = (bf16)ra[i];
    }
    #pragma unroll
    for (int i = 0; i < 8; ++i) BsT[b_n0 + i][b_kk] = (bf16)rb[i];
  };

  const int Kpad = (Ktrue + TKS - 1) / TKS * TKS;
  loadA(0); loadB(0);

  for (int k0 = 0; k0 < Kpad; k0 += TKS) {
    storeTiles();
    __syncthreads();
    if (k0 + TKS < Kpad) {                 // overlap next-slab loads with WMMAs
      loadA(k0 + TKS); loadB(k0 + TKS);
      if (k0 + 2 * TKS < Kpad) {           // distance-2 speculative prefetch
        const float* pf = (TRANSA == 0)
            ? A + (long)(tileM0 + a_r) * lda + (k0 + 2 * TKS + a_c0)
            : A + (long)(k0 + 2 * TKS + a_kk) * lda + (tileM0 + a_m0);
        __builtin_prefetch(pf, 0, 1);
      }
    }
    v16bf a0 = ldfrag(&As[wm + lm][0], hl);
    v16bf a1 = ldfrag(&As[wm + 16 + lm][0], hl);
    v16bf b0 = ldfrag(&BsT[wn + lm][0], hl);
    v16bf b1 = ldfrag(&BsT[wn + 16 + lm][0], hl);
    acc[0][0] = __builtin_amdgcn_wmma_f32_16x16x32_bf16(false, a0, false, b0, (short)0, acc[0][0], false, false);
    acc[0][1] = __builtin_amdgcn_wmma_f32_16x16x32_bf16(false, a0, false, b1, (short)0, acc[0][1], false, false);
    acc[1][0] = __builtin_amdgcn_wmma_f32_16x16x32_bf16(false, a1, false, b0, (short)0, acc[1][0], false, false);
    acc[1][1] = __builtin_amdgcn_wmma_f32_16x16x32_bf16(false, a1, false, b1, (short)0, acc[1][1], false, false);
    __syncthreads();
  }

  // Epilogue. C/D layout: lanes 0-15 -> M=r, lanes 16-31 -> M=r+8; N = lane%16.
  #pragma unroll
  for (int dn = 0; dn < 2; ++dn) {
    const int gn = tileN0 + wn + dn * 16 + lm;
    if (gn >= Ntot) continue;
    const float bv = bias ? bias[gn] : 0.f;
    #pragma unroll
    for (int dm = 0; dm < 2; ++dm) {
      const v8f a8 = acc[dm][dn];
      const int mb = tileM0 + wm + dm * 16 + hl * 8;
      #pragma unroll
      for (int r = 0; r < 8; ++r) {
        const int gm = mb + r;
        if (gm >= Mtot) continue;
        float y = a8[r] + bv;
        if (EPI == EPI_LIN) {
          Cp[(long)gm * Ntot + gn] = y;
        } else if (EPI == EPI_ELU) {
          Cp[(long)gm * Ntot + gn] = eluf(y);
        } else if (EPI == EPI_MASK) {
          Cp[(long)gm * Ntot + gn] = y * mask[gm];
        } else {  // scatter (b,h,n,k) layout for q/k/v projections
          const int b = gm >> 12, n = gm & (NN - 1);
          const int h = gn >> 7,  kc = gn & 127;
          const long o = (((long)(b * HH + h) * NN + n) << 7) + kc;
          if (EPI == EPI_QP)      Cp[o] = eluf(y * 0.088388347648318447f) + 1.f;  // K^-1/2
          else if (EPI == EPI_KP) Cp[o] = (eluf(y) + 1.f) * mask[gm];
          else                    Cp[o] = y;
        }
      }
    }
  }
}

// ----------------------------- LayerNorm -----------------------------------
__global__ __launch_bounds__(128) void ln_kernel(
    const float* __restrict__ Xp, const float* __restrict__ Radd,
    const float* __restrict__ g, const float* __restrict__ b,
    float* __restrict__ Y)
{
  int m = blockIdx.x, t = threadIdx.x;
  float x = Xp[(long)m * DD + t];
  if (Radd) x += Radd[(long)m * DD + t];
  __shared__ float red[DD];
  red[t] = x; __syncthreads();
  for (int s = 64; s > 0; s >>= 1) { if (t < s) red[t] += red[t + s]; __syncthreads(); }
  float mean = red[0] * (1.f / DD); __syncthreads();
  float d = x - mean;
  red[t] = d * d; __syncthreads();
  for (int s = 64; s > 0; s >>= 1) { if (t < s) red[t] += red[t + s]; __syncthreads(); }
  float var = red[0] * (1.f / DD);
  Y[(long)m * DD + t] = d * rsqrtf(var + 1e-3f) * g[t] + b[t];
}

// -------------------------- attention helpers ------------------------------
__global__ __launch_bounds__(128) void ksum_kernel(const float* __restrict__ KP,
                                                   float* __restrict__ KS)
{
  int bh = blockIdx.x, k = threadIdx.x;
  const float* p = KP + (long)bh * NN * DD + k;
  float s = 0.f;
  for (int n = 0; n < NN; ++n) s += p[(long)n * DD];
  KS[bh * DD + k] = s;
}

__global__ void z_kernel(const float* __restrict__ QP, const float* __restrict__ KS,
                         float* __restrict__ Z, int total)
{
  int i = blockIdx.x * blockDim.x + threadIdx.x;
  if (i >= total) return;
  int m = i >> 3, h = i & 7;
  int b = m >> 12, n = m & (NN - 1);
  const float4* q  = (const float4*)(QP + ((long)(b * HH + h) * NN + n) * DD);
  const float4* ks = (const float4*)(KS + (b * HH + h) * DD);
  float s = 0.f;
  for (int k = 0; k < DD / 4; ++k) {
    float4 a = q[k], c = ks[k];
    s += a.x * c.x + a.y * c.y + a.z * c.z + a.w * c.w;
  }
  Z[i] = s;
}

__global__ void outf_kernel(const float* __restrict__ NUMp, const float* __restrict__ Z,
                            float* __restrict__ OUTF, int total)
{
  int i = blockIdx.x * blockDim.x + threadIdx.x;
  if (i >= total) return;
  int m = i >> 10, j = i & 1023;
  int h = j >> 7, v = j & 127;
  int b = m >> 12, n = m & (NN - 1);
  long src = (((long)(b * HH + h) * NN + n) << 7) + v;
  OUTF[i] = NUMp[src] / (Z[(m << 3) + h] + 1e-6f);
}

// -------------------------- small dense / misc -----------------------------
__global__ void dense_small(const float* __restrict__ A, const float* __restrict__ W,
                            const float* __restrict__ bias, float* __restrict__ C,
                            const float* __restrict__ mask,
                            int Mtot, int Kd, int Nout, int act, int useMask)
{
  int m = blockIdx.x * blockDim.x + threadIdx.x;
  if (m >= Mtot) return;
  const float* a = A + (long)m * Kd;
  float mm = useMask ? mask[m] : 1.f;
  for (int j = 0; j < Nout; ++j) {
    float s = bias ? bias[j] : 0.f;
    for (int k = 0; k < Kd; ++k) s += a[k] * W[k * Nout + j];
    if (act) s = eluf(s);
    C[(long)m * Nout + j] = s * mm;
  }
}

__global__ void softmax8(const float* __restrict__ L, float* __restrict__ C, int Mtot)
{
  int m = blockIdx.x * blockDim.x + threadIdx.x;
  if (m >= Mtot) return;
  const float* l = L + (long)m * 8;
  float* c = C + (long)m * 8;
  float mx = l[0];
  for (int i = 1; i < 8; ++i) mx = fmaxf(mx, l[i]);
  float e[8], s = 0.f;
  for (int i = 0; i < 8; ++i) { e[i] = expf(l[i] - mx); s += e[i]; }
  float inv = 1.f / s;
  for (int i = 0; i < 8; ++i) c[i] = e[i] * inv;
}

// ------------------------------ encoding -----------------------------------
__global__ void encode_kernel(const float* __restrict__ X, float* __restrict__ XF,
                              float* __restrict__ mask, int Mtot)
{
  int m = blockIdx.x * blockDim.x + threadIdx.x;
  if (m >= Mtot) return;
  const float* x = X + (long)m * 17;
  float id = x[0];
  mask[m] = (id != 0.f) ? 1.f : 0.f;
  float* o = XF + (long)m * 64;
  int c = (int)id;
  for (int i = 0; i < 8; ++i) o[i] = (i == c) ? 1.f : 0.f;
  float pt  = logf(x[1] + 1.f);
  float e   = logf(x[4] + 1.f);
  float eta = x[2];
  o[8]  = pt; o[9] = sqrtf(pt); o[10] = pt * pt;
  o[11] = fminf(fmaxf(sinhf(eta), -10.f), 10.f);
  o[12] = fminf(fmaxf(coshf(eta), -10.f), 10.f);
  o[13] = fabsf(eta);
  o[14] = sinf(x[3]); o[15] = cosf(x[3]);
  o[16] = e; o[17] = sqrtf(e); o[18] = e * e;
  o[19] = sinf(x[10]); o[20] = cosf(x[10]);
  o[21] = sinf(x[12]); o[22] = cosf(x[12]);
  for (int i = 0; i < 17; ++i) o[23 + i] = x[i];
  for (int i = 40; i < 64; ++i) o[i] = 0.f;   // K pad
}

__global__ void build_xc(const float* __restrict__ Xi, const float* __restrict__ LG,
                         float* __restrict__ XC, int Mtot)
{
  int m = blockIdx.x * blockDim.x + threadIdx.x;
  if (m >= Mtot) return;
  float* o = XC + (long)m * 160;
  const float* xi = Xi + (long)m * 128;
  for (int i = 0; i < 128; ++i) o[i] = xi[i];
  const float* lg = LG + (long)m * 8;
  for (int i = 0; i < 8; ++i) o[128 + i] = lg[i];
  for (int i = 136; i < 160; ++i) o[i] = 0.f;
}

__global__ void build_xce(const float* __restrict__ XC, const float* __restrict__ Xr,
                          const float* __restrict__ LG, float* __restrict__ XE, int Mtot)
{
  int m = blockIdx.x * blockDim.x + threadIdx.x;
  if (m >= Mtot) return;
  float* o = XE + (long)m * 288;
  const float* xc = XC + (long)m * 160;
  for (int i = 0; i < 136; ++i) o[i] = xc[i];
  const float* xr = Xr + (long)m * 128;
  for (int i = 0; i < 128; ++i) o[136 + i] = xr[i];
  const float* lg = LG + (long)m * 8;
  for (int i = 0; i < 8; ++i) o[264 + i] = lg[i];
  for (int i = 272; i < 288; ++i) o[i] = 0.f;
}

__global__ void assemble_kernel(const float* __restrict__ X, const float* __restrict__ mask,
                                const float* __restrict__ CLS, const float* __restrict__ CHG,
                                const float* __restrict__ ETA, const float* __restrict__ PHI,
                                const float* __restrict__ EC,  const float* __restrict__ PT,
                                float* __restrict__ out, int Mtot)
{
  int m = blockIdx.x * blockDim.x + threadIdx.x;
  if (m >= Mtot) return;
  float m1 = mask[m];
  const float* x = X + (long)m * 17;
  float* o = out + (long)m * 14;
  for (int j = 0; j < 8; ++j) o[j] = CLS[(long)m * 8 + j];
  o[8] = CHG[m];
  float ppt = fabsf(x[1] * PT[m * 2 + 0] + PT[m * 2 + 1]);
  o[9]  = ppt * m1;
  o[10] = (x[2] + ETA[m]) * m1;
  o[11] = (sinf(x[3]) * m1 + PHI[m * 2 + 0]) * m1;
  o[12] = (cosf(x[3]) * m1 + PHI[m * 2 + 1]) * m1;
  o[13] = fabsf(x[4] * m1 + EC[m]) * m1;
}

// ------------------------------- host side ---------------------------------
struct EncP {
  const float *Wk, *Wo, *Wq, *Wv, *b1, *b2, *bk, *bo, *bq, *bv;
  const float *fW0, *fb0, *fW1, *fb1, *g1, *g2;
};

extern "C" void kernel_launch(void* const* d_in, const int* in_sizes, int n_in,
                              void* d_out, int out_size, void* d_ws, size_t ws_size,
                              hipStream_t stream)
{
  (void)in_sizes; (void)n_in; (void)out_size; (void)ws_size;
  int ii = 0;
  auto nxt = [&]() { return (const float*)d_in[ii++]; };

  const float* X = nxt();
  // params['embed'] : [(W(40,128),b),(W(128,128),b)]
  const float* embW0 = nxt(); const float* embB0 = nxt();
  const float* embW1 = nxt(); const float* embB1 = nxt();
  // enc_id[3] then enc_reg[3]; per-layer key order (sorted):
  // Wk, Wo, Wq, Wv, b1, b2, bk, bo, bq, bv, ffn(W0,b0,W1,b1), g1, g2
  EncP enc[6];
  for (int l = 0; l < 6; ++l) {
    EncP& p = enc[l];
    p.Wk = nxt(); p.Wo = nxt(); p.Wq = nxt(); p.Wv = nxt();
    p.b1 = nxt(); p.b2 = nxt(); p.bk = nxt(); p.bo = nxt();
    p.bq = nxt(); p.bv = nxt();
    p.fW0 = nxt(); p.fb0 = nxt(); p.fW1 = nxt(); p.fb1 = nxt();
    p.g1 = nxt(); p.g2 = nxt();
  }
  // out heads, sorted: charge(3), energy(4), eta(4), id(5), phi(4), pt(4)
  const float *chW[3], *chB[3]; for (int i = 0; i < 3; ++i) { chW[i] = nxt(); chB[i] = nxt(); }
  const float *enW[4], *enB[4]; for (int i = 0; i < 4; ++i) { enW[i] = nxt(); enB[i] = nxt(); }
  const float *etW[4], *etB[4]; for (int i = 0; i < 4; ++i) { etW[i] = nxt(); etB[i] = nxt(); }
  const float *idW[5], *idB[5]; for (int i = 0; i < 5; ++i) { idW[i] = nxt(); idB[i] = nxt(); }
  const float *phW[4], *phB[4]; for (int i = 0; i < 4; ++i) { phW[i] = nxt(); phB[i] = nxt(); }
  const float *ptW[4], *ptB[4]; for (int i = 0; i < 4; ++i) { ptW[i] = nxt(); ptB[i] = nxt(); }

  // workspace carve (floats)
  float* ws = (float*)d_ws;
  size_t off = 0;
  auto carve = [&](size_t n) { float* p = ws + off; off += n; return p; };
  float* MSK  = carve(MTOK);
  float* XF   = carve((size_t)MTOK * 64);
  float* T0   = carve((size_t)MTOK * DD);
  float* T1   = carve((size_t)MTOK * DD);
  float* Ebuf = carve((size_t)MTOK * DD);
  float* Xi   = carve((size_t)MTOK * DD);
  float* Xr   = carve((size_t)MTOK * DD);
  float* Xn   = carve((size_t)MTOK * DD);
  float* ATT  = carve((size_t)MTOK * DD);
  float* QP   = carve((size_t)MTOK * HK);   // reused as OUTF
  float* KP   = carve((size_t)MTOK * HK);   // reused as NUM
  float* VV   = carve((size_t)MTOK * HK);
  float* KV   = carve((size_t)32 * DD * DD);
  float* KS   = carve((size_t)32 * DD);
  float* Z    = carve((size_t)MTOK * HH);
  float* LG   = carve((size_t)MTOK * 8);
  float* CLS  = carve((size_t)MTOK * 8);
  float* CHG  = carve(MTOK);
  float* ETA  = carve(MTOK);
  float* PHI  = carve((size_t)MTOK * 2);
  float* EC   = carve(MTOK);
  float* PT   = carve((size_t)MTOK * 2);
  float* XC   = carve((size_t)MTOK * 160);
  float* XE   = carve((size_t)MTOK * 288);

  const int TB = 256, NB = (MTOK + TB - 1) / TB;

  auto gemm = [&](const float* A, const float* Bw, const float* bias, float* C,
                  int M_, int N_, int Kt, int lda, int transA,
                  long sA, long sB, long sC, int nb, int epi) {
    dim3 grid((N_ + TN - 1) / TN, (M_ + TM - 1) / TM, nb);
    #define GG(TA, EP) wmma_gemm<TA, EP><<<grid, TB, 0, stream>>>( \
        A, Bw, bias, C, MSK, M_, N_, Kt, lda, sA, sB, sC)
    if (transA) { GG(1, EPI_LIN); }
    else switch (epi) {
      case EPI_LIN:  GG(0, EPI_LIN);  break;
      case EPI_ELU:  GG(0, EPI_ELU);  break;
      case EPI_QP:   GG(0, EPI_QP);   break;
      case EPI_KP:   GG(0, EPI_KP);   break;
      case EPI_SCAT: GG(0, EPI_SCAT); break;
      default:       GG(0, EPI_MASK); break;
    }
    #undef GG
  };

  auto run_enc = [&](const float* Xq, float* Xout, const EncP& p) {
    ln_kernel<<<MTOK, 128, 0, stream>>>(Xq, nullptr, p.g1, p.b1, Xn);
    // q/k/v projections: 16384x1024x128 GEMMs, fused activation + (b,h,n,k) scatter
    gemm(Xq, p.Wq, p.bq, QP, MTOK, HK, DD, DD, 0, 0, 0, 0, 1, EPI_QP);
    gemm(Xn, p.Wk, p.bk, KP, MTOK, HK, DD, DD, 0, 0, 0, 0, 1, EPI_KP);
    gemm(Xn, p.Wv, p.bv, VV, MTOK, HK, DD, DD, 0, 0, 0, 0, 1, EPI_SCAT);
    ksum_kernel<<<32, 128, 0, stream>>>(KP, KS);
    // kv[b,h] = kp^T (128x4096) x v (4096x128) : batched WMMA GEMM
    gemm(KP, VV, nullptr, KV, DD, DD, NN, DD, 1,
         (long)NN * DD, (long)NN * DD, (long)DD * DD, 32, EPI_LIN);
    z_kernel<<<(MTOK * HH + TB - 1) / TB, TB, 0, stream>>>(QP, KS, Z, MTOK * HH);
    // num[b,h] = qp (4096x128) x kv (128x128), NUM reuses KP storage
    gemm(QP, KV, nullptr, KP, NN, DD, DD, DD, 0,
         (long)NN * DD, (long)DD * DD, (long)NN * DD, 32, EPI_LIN);
    // normalize and flatten to (m, h*v); OUTF reuses QP storage
    outf_kernel<<<(MTOK * HK) / TB, TB, 0, stream>>>(KP, Z, QP, MTOK * HK);
    gemm(QP, p.Wo, p.bo, ATT, MTOK, DD, HK, HK, 0, 0, 0, 0, 1, EPI_MASK);
    ln_kernel<<<MTOK, 128, 0, stream>>>(Xn, ATT, p.g2, p.b2, T0);
    gemm(T0, p.fW0, p.fb0, T1, MTOK, DD, DD, DD, 0, 0, 0, 0, 1, EPI_ELU);
    gemm(T1, p.fW1, p.fb1, Xout, MTOK, DD, DD, DD, 0, 0, 0, 0, 1, EPI_LIN);
  };

  // ---- input encoding + embed MLP ----
  encode_kernel<<<NB, TB, 0, stream>>>(X, XF, MSK, MTOK);
  gemm(XF, embW0, embB0, T0, MTOK, DD, 40, 64, 0, 0, 0, 0, 1, EPI_ELU);
  gemm(T0, embW1, embB1, Ebuf, MTOK, DD, DD, DD, 0, 0, 0, 0, 1, EPI_LIN);

  // ---- encoder stacks ----
  run_enc(Ebuf, Xi, enc[0]); run_enc(Xi, Xi, enc[1]); run_enc(Xi, Xi, enc[2]);
  run_enc(Ebuf, Xr, enc[3]); run_enc(Xr, Xr, enc[4]); run_enc(Xr, Xr, enc[5]);

  // ---- ffn_id (128->128->64->32->16->8) + softmax ----
  gemm(Xi, idW[0], idB[0], T0, MTOK, 128, 128, 128, 0, 0, 0, 0, 1, EPI_ELU);
  gemm(T0, idW[1], idB[1], T1, MTOK, 64, 128, 128, 0, 0, 0, 0, 1, EPI_ELU);
  gemm(T1, idW[2], idB[2], T0, MTOK, 32, 64, 64, 0, 0, 0, 0, 1, EPI_ELU);
  dense_small<<<NB, TB, 0, stream>>>(T0, idW[3], idB[3], T1, MSK, MTOK, 32, 16, 1, 0);
  dense_small<<<NB, TB, 0, stream>>>(T1, idW[4], idB[4], LG, MSK, MTOK, 16, 8, 0, 1);
  softmax8<<<NB, TB, 0, stream>>>(LG, CLS, MTOK);

  // ---- ffn_charge (128->128->64->1) ----
  gemm(Xi, chW[0], chB[0], T0, MTOK, 128, 128, 128, 0, 0, 0, 0, 1, EPI_ELU);
  gemm(T0, chW[1], chB[1], T1, MTOK, 64, 128, 128, 0, 0, 0, 0, 1, EPI_ELU);
  dense_small<<<NB, TB, 0, stream>>>(T1, chW[2], chB[2], CHG, MSK, MTOK, 64, 1, 0, 1);

  // ---- Xc = [Xi, logits] (136, padded to 160) ----
  build_xc<<<NB, TB, 0, stream>>>(Xi, LG, XC, MTOK);

  // ---- ffn_eta ----
  gemm(XC, etW[0], etB[0], T0, MTOK, 128, 136, 160, 0, 0, 0, 0, 1, EPI_ELU);
  gemm(T0, etW[1], etB[1], T1, MTOK, 128, 128, 128, 0, 0, 0, 0, 1, EPI_ELU);
  gemm(T1, etW[2], etB[2], T0, MTOK, 128, 128, 128, 0, 0, 0, 0, 1, EPI_ELU);
  dense_small<<<NB, TB, 0, stream>>>(T0, etW[3], etB[3], ETA, MSK, MTOK, 128, 1, 0, 1);

  // ---- ffn_phi ----
  gemm(XC, phW[0], phB[0], T0, MTOK, 128, 136, 160, 0, 0, 0, 0, 1, EPI_ELU);
  gemm(T0, phW[1], phB[1], T1, MTOK, 128, 128, 128, 0, 0, 0, 0, 1, EPI_ELU);
  gemm(T1, phW[2], phB[2], T0, MTOK, 128, 128, 128, 0, 0, 0, 0, 1, EPI_ELU);
  dense_small<<<NB, TB, 0, stream>>>(T0, phW[3], phB[3], PHI, MSK, MTOK, 128, 2, 0, 1);

  // ---- Xce = [Xc, Xr, logits] (272, padded to 288) ----
  build_xce<<<NB, TB, 0, stream>>>(XC, Xr, LG, XE, MTOK);

  // ---- ffn_energy ----
  gemm(XE, enW[0], enB[0], T0, MTOK, 128, 272, 288, 0, 0, 0, 0, 1, EPI_ELU);
  gemm(T0, enW[1], enB[1], T1, MTOK, 128, 128, 128, 0, 0, 0, 0, 1, EPI_ELU);
  gemm(T1, enW[2], enB[2], T0, MTOK, 128, 128, 128, 0, 0, 0, 0, 1, EPI_ELU);
  dense_small<<<NB, TB, 0, stream>>>(T0, enW[3], enB[3], EC, MSK, MTOK, 128, 1, 0, 1);

  // ---- ffn_pt ----
  gemm(XE, ptW[0], ptB[0], T0, MTOK, 128, 272, 288, 0, 0, 0, 0, 1, EPI_ELU);
  gemm(T0, ptW[1], ptB[1], T1, MTOK, 128, 128, 128, 0, 0, 0, 0, 1, EPI_ELU);
  gemm(T1, ptW[2], ptB[2], T0, MTOK, 128, 128, 128, 0, 0, 0, 0, 1, EPI_ELU);
  dense_small<<<NB, TB, 0, stream>>>(T0, ptW[3], ptB[3], PT, MSK, MTOK, 128, 2, 0, 1);

  // ---- final assembly: (B,N,14) ----
  assemble_kernel<<<NB, TB, 0, stream>>>(X, MSK, CLS, CHG, ETA, PHI, EC, PT,
                                         (float*)d_out, MTOK);
}
